// ConditionalMamba2DBlock_72155450573522
// MI455X (gfx1250) — compile-verified, matrix operand order checked by reading, not run on hardware
//
#include <hip/hip_runtime.h>
#include <hip/hip_bf16.h>
#include <math.h>

// ---------------------------------------------------------------------------
// Problem constants (from the reference)
// ---------------------------------------------------------------------------
#define BB        8
#define D_MODEL   256
#define TT        256
#define FREQ      64
#define CTX_DIM   128
#define D_STATE   16
#define D_CONV    4
#define D_INNER   512
#define HEADDIM   64
#define NHEADS    8
#define CONV_DIM  544           // D_INNER + 2*D_STATE
#define D_IN_PROJ 1064          // 2*D_INNER + 2*D_STATE + NHEADS
#define NSEQ      (BB * FREQ)   // 512 sequences
#define NROWS     (NSEQ * TT)   // 131072 rows through the GEMMs
#define EPSF      1e-5f

typedef __attribute__((ext_vector_type(16))) _Float16 v16h;
typedef __attribute__((ext_vector_type(8)))  _Float16 v8h;
typedef __attribute__((ext_vector_type(8)))  float    v8f;

// ---------------------------------------------------------------------------
// Kernel 1: transpose u [B,C,T,F] -> x [bf= b*F+f, t, c], LayerNorm over c,
// emit fp16 for the WMMA GEMM.
// ---------------------------------------------------------------------------
__global__ void k_transpose_ln(const float* __restrict__ u,
                               const float* __restrict__ lnw,
                               const float* __restrict__ lnb,
                               _Float16* __restrict__ xln) {
  const int row = blockIdx.x;           // bf*TT + t
  const int bf  = row >> 8;
  const int t   = row & (TT - 1);
  const int b   = bf >> 6;
  const int f   = bf & (FREQ - 1);
  const int c   = threadIdx.x;          // 256 threads

  const float v = u[(((size_t)(b * D_MODEL + c) * TT) + t) * FREQ + f];

  __shared__ float s1[256], s2[256];
  s1[c] = v;
  s2[c] = v * v;
  __syncthreads();
  #pragma unroll
  for (int s = 128; s > 0; s >>= 1) {
    if (c < s) { s1[c] += s1[c + s]; s2[c] += s2[c + s]; }
    __syncthreads();
  }
  const float mu  = s1[0] * (1.f / 256.f);
  const float var = s2[0] * (1.f / 256.f) - mu * mu;
  const float inv = rsqrtf(var + EPSF);
  xln[(size_t)row * D_MODEL + c] = (_Float16)((v - mu) * inv * lnw[c] + lnb[c]);
}

// ---------------------------------------------------------------------------
// Kernel: pack a row-major fp32 weight [K, N] into WMMA B-fragment order.
// For v_wmma_f32_16x16x32_f16, lane L of a 32x16 B tile holds column
// n = (L&15) and 16 halves at k = base0 + j + (j>=8 ? 8 : 0), base0 = 8*(L>=16).
// Packed layout: Bp[((kt*ntiles + nt) * 32 + lane) * 16 + j]  (32B/lane, coalesced)
// ---------------------------------------------------------------------------
__global__ void k_pack_b(const float* __restrict__ W, _Float16* __restrict__ Bp,
                         int K, int N, int ktiles, int ntiles, int ldw) {
  const int idx = blockIdx.x * blockDim.x + threadIdx.x;   // one per (tile, lane)
  if (idx >= ktiles * ntiles * 32) return;
  const int lane = idx & 31;
  const int tile = idx >> 5;
  const int nt = tile % ntiles;
  const int kt = tile / ntiles;
  const int half  = lane & 15;
  const int base0 = (lane >> 4) * 8;
  const int n = nt * 16 + half;
  _Float16* dst = Bp + ((size_t)tile << 9) + lane * 16;
  #pragma unroll
  for (int j = 0; j < 16; ++j) {
    const int k = kt * 32 + base0 + j + ((j >= 8) ? 8 : 0);
    const float v = (k < K && n < N) ? W[(size_t)k * ldw + n] : 0.f;
    dst[j] = (_Float16)v;
  }
}

// ---------------------------------------------------------------------------
// WMMA GEMM: C[M,N] = A[M,K] (fp16 row-major) * Bpacked (fp16 fragments),
// fp32 accumulate via v_wmma_f32_16x16x32_f16.
//
// Each wave owns a 32x64 output tile (2 M-tiles x 4 N-tiles, 8 accumulators)
// with an explicitly software-pipelined K-loop (fragments for kt+1 issued
// before the 8 WMMAs of kt). 128-thread blocks + launch_bounds(128,1) give
// one wave per SIMD the full VGPR file so the ~190-register live set does
// not spill; latency is hidden by the pipeline, parallelism by 17K blocks.
// ntiles must be a multiple of 4 (packed B is zero-padded).
// ---------------------------------------------------------------------------
__device__ inline v16h load_a16(const _Float16* __restrict__ p) {
  const v8h lo = *reinterpret_cast<const v8h*>(p);        // k = base0 .. base0+7
  const v8h hi = *reinterpret_cast<const v8h*>(p + 16);   // k = base0+16 .. +23
  v16h r;
  #pragma unroll
  for (int i = 0; i < 8; ++i) { r[i] = lo[i]; r[i + 8] = hi[i]; }
  return r;
}

__device__ inline void store_tile(float* __restrict__ C, int ldc, const v8f acc,
                                  int m0, int n0, int Nstore) {
  const int lane = threadIdx.x & 31;
  const int n = n0 + (lane & 15);
  if (n >= Nstore) return;
  const int hi8 = (lane >> 4) * 8;
  #pragma unroll
  for (int r = 0; r < 8; ++r)
    C[(size_t)(m0 + r + hi8) * ldc + n] = acc[r];
}

#define WMMA_F16(a, b, c) \
  __builtin_amdgcn_wmma_f32_16x16x32_f16(false, (a), false, (b), (short)0, (c), false, false)

__global__ __launch_bounds__(128, 1)
void k_wmma_gemm(const _Float16* __restrict__ A,
                 const _Float16* __restrict__ Bp,
                 float* __restrict__ C,
                 int mgroups, int Nstore, int K,
                 int lda, int ldc, int ntiles) {
  const int wave    = blockIdx.x * (blockDim.x >> 5) + (threadIdx.x >> 5);
  const int ngroups = ntiles >> 2;
  const int mg = wave / ngroups;
  const int ng = wave % ngroups;
  if (mg >= mgroups) return;
  const int m0  = mg * 32;            // two 16-row tiles: m0, m0+16
  const int nt0 = ng * 4;

  const int lane  = threadIdx.x & 31;
  const int half  = lane & 15;
  const int base0 = (lane >> 4) * 8;

  // running pointers -> base + immediate-offset b128 loads in the loop
  const _Float16* ap0 = A + (size_t)(m0 + half) * lda + base0;   // M-tile 0
  const _Float16* ap1 = ap0 + (size_t)16 * lda;                  // M-tile 1
  const _Float16* bp  = Bp + ((size_t)nt0 << 9) + lane * 16;     // 4 N-tiles
  const size_t    bstep = (size_t)ntiles << 9;                   // halves per kt

  v8f acc00 = {}, acc01 = {}, acc02 = {}, acc03 = {};
  v8f acc10 = {}, acc11 = {}, acc12 = {}, acc13 = {};

  const int ksteps = K >> 5;

  // prologue: fragments for kt = 0
  v16h a0 = load_a16(ap0);
  v16h a1 = load_a16(ap1);
  v16h b0 = *reinterpret_cast<const v16h*>(bp);
  v16h b1 = *reinterpret_cast<const v16h*>(bp + 512);
  v16h b2 = *reinterpret_cast<const v16h*>(bp + 1024);
  v16h b3 = *reinterpret_cast<const v16h*>(bp + 1536);

  for (int kt = 0; kt < ksteps - 1; ++kt) {
    ap0 += 32; ap1 += 32; bp += bstep;
    // issue next step's loads first so they overlap this step's WMMAs
    const v16h na0 = load_a16(ap0);
    const v16h na1 = load_a16(ap1);
    const v16h nb0 = *reinterpret_cast<const v16h*>(bp);
    const v16h nb1 = *reinterpret_cast<const v16h*>(bp + 512);
    const v16h nb2 = *reinterpret_cast<const v16h*>(bp + 1024);
    const v16h nb3 = *reinterpret_cast<const v16h*>(bp + 1536);

    acc00 = WMMA_F16(a0, b0, acc00);
    acc01 = WMMA_F16(a0, b1, acc01);
    acc02 = WMMA_F16(a0, b2, acc02);
    acc03 = WMMA_F16(a0, b3, acc03);
    acc10 = WMMA_F16(a1, b0, acc10);
    acc11 = WMMA_F16(a1, b1, acc11);
    acc12 = WMMA_F16(a1, b2, acc12);
    acc13 = WMMA_F16(a1, b3, acc13);

    a0 = na0; a1 = na1;
    b0 = nb0; b1 = nb1; b2 = nb2; b3 = nb3;
  }

  // epilogue: last K-step
  acc00 = WMMA_F16(a0, b0, acc00);
  acc01 = WMMA_F16(a0, b1, acc01);
  acc02 = WMMA_F16(a0, b2, acc02);
  acc03 = WMMA_F16(a0, b3, acc03);
  acc10 = WMMA_F16(a1, b0, acc10);
  acc11 = WMMA_F16(a1, b1, acc11);
  acc12 = WMMA_F16(a1, b2, acc12);
  acc13 = WMMA_F16(a1, b3, acc13);

  store_tile(C, ldc, acc00, m0,      (nt0 + 0) * 16, Nstore);
  store_tile(C, ldc, acc01, m0,      (nt0 + 1) * 16, Nstore);
  store_tile(C, ldc, acc02, m0,      (nt0 + 2) * 16, Nstore);
  store_tile(C, ldc, acc03, m0,      (nt0 + 3) * 16, Nstore);
  store_tile(C, ldc, acc10, m0 + 16, (nt0 + 0) * 16, Nstore);
  store_tile(C, ldc, acc11, m0 + 16, (nt0 + 1) * 16, Nstore);
  store_tile(C, ldc, acc12, m0 + 16, (nt0 + 2) * 16, Nstore);
  store_tile(C, ldc, acc13, m0 + 16, (nt0 + 3) * 16, Nstore);
}

// ---------------------------------------------------------------------------
// Kernel: causal depthwise conv (k=4) over t on xBC channels + SiLU.
// zxbcdt layout per row (bf*TT + t): [z 0..511 | xBC 512..1055 | dt 1056..1063]
// ---------------------------------------------------------------------------
__global__ void k_conv_silu(const float* __restrict__ zx,
                            const float* __restrict__ cw,
                            const float* __restrict__ cb,
                            float* __restrict__ xconv) {
  const size_t idx = (size_t)blockIdx.x * blockDim.x + threadIdx.x;
  if (idx >= (size_t)NROWS * CONV_DIM) return;
  const int    c   = (int)(idx % CONV_DIM);
  const size_t row = idx / CONV_DIM;                 // bf*TT + t
  const int    t   = (int)(row & (TT - 1));
  const size_t base = (row - t) * D_IN_PROJ;         // start of this sequence
  float acc = cb[c];
  #pragma unroll
  for (int k = 0; k < D_CONV; ++k) {
    const int tt = t - (D_CONV - 1) + k;
    if (tt >= 0)
      acc += cw[c * D_CONV + k] * zx[base + (size_t)tt * D_IN_PROJ + D_INNER + c];
  }
  xconv[row * CONV_DIM + c] = acc / (1.f + __expf(-acc));   // SiLU
}

// ---------------------------------------------------------------------------
// Kernel: Mamba2 selective-state scan. One block per (sequence, head);
// 64 threads, each owning one P-row of the 64x16 state in registers.
// ---------------------------------------------------------------------------
__global__ void k_ssm_scan(const float* __restrict__ zx,
                           const float* __restrict__ xconv,
                           const float* __restrict__ dt_bias,
                           const float* __restrict__ A_log,
                           const float* __restrict__ Dp,
                           float* __restrict__ ybuf) {
  const int blk = blockIdx.x;          // bf*NHEADS + h
  const int h  = blk & (NHEADS - 1);
  const int bf = blk >> 3;
  const int p  = threadIdx.x;          // 0..63

  float st[D_STATE];
  #pragma unroll
  for (int n = 0; n < D_STATE; ++n) st[n] = 0.f;

  __shared__ float sB[D_STATE], sC[D_STATE], sdt, sdA;
  const float expA = __expf(A_log[h]);
  const float dtb  = dt_bias[h];
  const float Dh   = Dp[h];

  for (int t = 0; t < TT; ++t) {
    const size_t row = (size_t)bf * TT + t;
    if (p < 16) {
      sB[p] = xconv[row * CONV_DIM + D_INNER + p];
    } else if (p < 32) {
      sC[p - 16] = xconv[row * CONV_DIM + D_INNER + D_STATE + (p - 16)];
    } else if (p == 32) {
      const float draw = zx[row * D_IN_PROJ + (D_IN_PROJ - NHEADS) + h] + dtb;
      const float dt = (draw > 20.f) ? draw : log1pf(__expf(draw));  // softplus
      sdt = dt;
      sdA = __expf(-expA * dt);
    }
    __syncthreads();
    const float dt  = sdt;
    const float dA  = sdA;
    const float xv  = xconv[row * CONV_DIM + h * HEADDIM + p];
    const float dtx = dt * xv;
    float acc = 0.f;
    #pragma unroll
    for (int n = 0; n < D_STATE; ++n) {
      st[n] = dA * st[n] + dtx * sB[n];
      acc  += st[n] * sC[n];
    }
    ybuf[row * D_INNER + h * HEADDIM + p] = acc + Dh * xv;
    __syncthreads();
  }
}

// ---------------------------------------------------------------------------
// Kernel: g = y * silu(z); gated RMSNorm over 512 channels; emit fp16 for GEMM2.
// ---------------------------------------------------------------------------
__global__ void k_gate_rms(const float* __restrict__ ybuf,
                           const float* __restrict__ zx,
                           const float* __restrict__ rmsw,
                           _Float16* __restrict__ g16) {
  const int row = blockIdx.x;          // bf*TT + t
  const int tid = threadIdx.x;         // 256 threads, 2 channels each
  float z0 = zx[(size_t)row * D_IN_PROJ + tid];
  float z1 = zx[(size_t)row * D_IN_PROJ + tid + 256];
  const float g0 = ybuf[(size_t)row * D_INNER + tid]       * (z0 / (1.f + __expf(-z0)));
  const float g1 = ybuf[(size_t)row * D_INNER + tid + 256] * (z1 / (1.f + __expf(-z1)));

  __shared__ float ss[256];
  ss[tid] = g0 * g0 + g1 * g1;
  __syncthreads();
  #pragma unroll
  for (int s = 128; s > 0; s >>= 1) {
    if (tid < s) ss[tid] += ss[tid + s];
    __syncthreads();
  }
  const float scale = rsqrtf(ss[0] * (1.f / (float)D_INNER) + EPSF);
  g16[(size_t)row * D_INNER + tid]       = (_Float16)(g0 * scale * rmsw[tid]);
  g16[(size_t)row * D_INNER + tid + 256] = (_Float16)(g1 * scale * rmsw[tid + 256]);
}

// ---------------------------------------------------------------------------
// Kernel: FiLM parameters gamma/beta[b, c] from the context vector (tiny).
// ---------------------------------------------------------------------------
__global__ void k_film(const float* __restrict__ ctx,
                       const float* __restrict__ gw, const float* __restrict__ gb,
                       const float* __restrict__ bw, const float* __restrict__ bbv,
                       float* __restrict__ gamma, float* __restrict__ beta) {
  const int idx = blockIdx.x * blockDim.x + threadIdx.x;
  if (idx >= BB * D_MODEL) return;
  const int b = idx >> 8, c = idx & (D_MODEL - 1);
  float ag = 0.f, ab = 0.f;
  for (int k = 0; k < CTX_DIM; ++k) {
    const float cv = ctx[b * CTX_DIM + k];
    ag += cv * gw[(size_t)k * D_MODEL + c];
    ab += cv * bw[(size_t)k * D_MODEL + c];
  }
  gamma[idx] = ag + gb[c];
  beta[idx]  = ab + bbv[c];
}

// ---------------------------------------------------------------------------
// Kernel: out[b,c,t,f] = u + gamma[b,c]*x[b,c,t,f] + beta[b,c]
// where x comes back from ymodel[(b*F+f)*T + t][c].
// ---------------------------------------------------------------------------
__global__ void k_final(const float* __restrict__ u,
                        const float* __restrict__ ymodel,
                        const float* __restrict__ gamma,
                        const float* __restrict__ beta,
                        float* __restrict__ out) {
  const size_t idx = (size_t)blockIdx.x * blockDim.x + threadIdx.x;
  if (idx >= (size_t)BB * D_MODEL * TT * FREQ) return;
  const int f = (int)(idx & (FREQ - 1));
  size_t r = idx >> 6;
  const int t = (int)(r & (TT - 1));  r >>= 8;
  const int c = (int)(r & (D_MODEL - 1));
  const int b = (int)(r >> 8);
  const float xm = ymodel[(((size_t)(b * FREQ + f) * TT) + t) * D_MODEL + c];
  const int bc = b * D_MODEL + c;
  out[idx] = u[idx] + gamma[bc] * xm + beta[bc];
}

// ---------------------------------------------------------------------------
// Launch
// ---------------------------------------------------------------------------
extern "C" void kernel_launch(void* const* d_in, const int* in_sizes, int n_in,
                              void* d_out, int out_size, void* d_ws, size_t ws_size,
                              hipStream_t stream) {
  const float* u        = (const float*)d_in[0];
  const float* ctx      = (const float*)d_in[1];
  const float* ln_w     = (const float*)d_in[2];
  const float* ln_b     = (const float*)d_in[3];
  const float* in_projw = (const float*)d_in[4];
  const float* conv_w   = (const float*)d_in[5];
  const float* conv_b   = (const float*)d_in[6];
  const float* dt_bias  = (const float*)d_in[7];
  const float* A_log    = (const float*)d_in[8];
  const float* Dp       = (const float*)d_in[9];
  const float* rms_w    = (const float*)d_in[10];
  const float* out_projw= (const float*)d_in[11];
  const float* gamma_w  = (const float*)d_in[12];
  const float* gamma_b  = (const float*)d_in[13];
  const float* beta_w   = (const float*)d_in[14];
  const float* beta_b   = (const float*)d_in[15];
  float* out = (float*)d_out;

  // GEMM tilings
  const int KT1 = D_MODEL / 32;          // 8 K-steps,  in_proj
  const int NT1 = 68;                    // ceil(1064/16)=67, padded to %4
  const int KT2 = D_INNER / 32;          // 16 K-steps, out_proj
  const int NT2 = D_MODEL / 16;          // 16
  const int MG  = NROWS / 32;            // 4096 32-row M-groups

  // Workspace carving (256B aligned)
  char* ws = (char*)d_ws;
  size_t off = 0;
  auto carve = [&](size_t bytes) { size_t o = off; off += (bytes + 255) & ~(size_t)255; return o; };
  _Float16* xln   = (_Float16*)(ws + carve((size_t)NROWS * D_MODEL * 2));
  _Float16* wpin  = (_Float16*)(ws + carve((size_t)KT1 * NT1 * 512 * 2));
  _Float16* wpout = (_Float16*)(ws + carve((size_t)KT2 * NT2 * 512 * 2));
  float*    zx    = (float*)   (ws + carve((size_t)NROWS * D_IN_PROJ * 4));
  float*    xconv = (float*)   (ws + carve((size_t)NROWS * CONV_DIM * 4));
  float*    ybuf  = (float*)   (ws + carve((size_t)NROWS * D_INNER * 4));
  _Float16* g16   = (_Float16*)(ws + carve((size_t)NROWS * D_INNER * 2));
  float*    ymod  = (float*)   (ws + carve((size_t)NROWS * D_MODEL * 4));
  float*    gamma = (float*)   (ws + carve((size_t)BB * D_MODEL * 4));
  float*    beta  = (float*)   (ws + carve((size_t)BB * D_MODEL * 4));
  (void)ws_size; (void)n_in; (void)in_sizes; (void)out_size;

  // 0) pack weights into WMMA fragment order (tiny)
  {
    int thr1 = KT1 * NT1 * 32;
    k_pack_b<<<(thr1 + 255) / 256, 256, 0, stream>>>(in_projw, wpin, D_MODEL, D_IN_PROJ, KT1, NT1, D_IN_PROJ);
    int thr2 = KT2 * NT2 * 32;
    k_pack_b<<<(thr2 + 255) / 256, 256, 0, stream>>>(out_projw, wpout, D_INNER, D_MODEL, KT2, NT2, D_MODEL);
  }

  // 1) transpose + LayerNorm -> fp16
  k_transpose_ln<<<NROWS, 256, 0, stream>>>(u, ln_w, ln_b, xln);

  // 2) in_proj GEMM (WMMA): zx = xln @ W_in   [131072 x 1064]
  {
    int waves  = MG * (NT1 / 4);                 // 4096 * 17
    int blocks = waves / 4;                      // 4 waves / 128-thread block
    k_wmma_gemm<<<blocks, 128, 0, stream>>>(xln, wpin, zx, MG, D_IN_PROJ, D_MODEL, D_MODEL, D_IN_PROJ, NT1);
  }

  // 3) causal depthwise conv + SiLU on xBC channels
  {
    size_t total = (size_t)NROWS * CONV_DIM;
    k_conv_silu<<<(unsigned)((total + 255) / 256), 256, 0, stream>>>(zx, conv_w, conv_b, xconv);
  }

  // 4) SSM scan: one block per (sequence, head)
  k_ssm_scan<<<NSEQ * NHEADS, HEADDIM, 0, stream>>>(zx, xconv, dt_bias, A_log, Dp, ybuf);

  // 5) gate + RMSNorm -> fp16
  k_gate_rms<<<NROWS, 256, 0, stream>>>(ybuf, zx, rms_w, g16);

  // 6) out_proj GEMM (WMMA): ymod = g @ W_out  [131072 x 256]
  {
    int waves  = MG * (NT2 / 4);                 // 4096 * 4
    int blocks = waves / 4;
    k_wmma_gemm<<<blocks, 128, 0, stream>>>(g16, wpout, ymod, MG, D_MODEL, D_INNER, D_INNER, D_MODEL, NT2);
  }

  // 7) FiLM params + final combine
  k_film<<<(BB * D_MODEL + 255) / 256, 256, 0, stream>>>(ctx, gamma_w, gamma_b, beta_w, beta_b, gamma, beta);
  {
    size_t total = (size_t)BB * D_MODEL * TT * FREQ;
    k_final<<<(unsigned)((total + 255) / 256), 256, 0, stream>>>(u, ymod, gamma, beta, out);
  }
}
